// LoRA_Moe_DepwiseConv_Samqv_1520418423068
// MI455X (gfx1250) — matrix operand order repaired and checked by reading, not compiled, not run on hardware
//
#include <hip/hip_runtime.h>

// ---------------------------------------------------------------------------
// CDNA5 (gfx1250) fused LoRA-MoE QKV projection.
// Pipeline:
//   k_f2bf4      : f32 -> bf16 conversion of W, lora_a_q, lora_a_v
//   k_pad_lorab  : lora_b [768,16] -> bf16 [768,32] zero-padded in K
//   k_zero       : clear gate accumulators
//   k_xconv_lora : x -> bf16 copy + d_q/d_v = x @ lora_a^T via WMMA bf16,
//                  plus atomic gate-mean partials
//   k_gate       : softmax gates (16 batches x 3 experts, q and v)
//   k_experts    : depthwise(3/5/7) + pointwise experts, prob-mixed, ->bf16
//   k_qkv_gemm   : main 65536x2304x768 WMMA bf16 GEMM with double-buffered
//                  GLOBAL_LOAD_ASYNC_TO_LDS staging of W tiles, bias init,
//                  fused +dq/+dv epilogue WMMA (K=16 padded to 32), and
//                  LDS-transposed b128 output stores.
// ---------------------------------------------------------------------------

typedef __attribute__((ext_vector_type(16))) __bf16 v16bf;
typedef __attribute__((ext_vector_type(8)))  __bf16 v8bf;
typedef __attribute__((ext_vector_type(8)))  float  v8f;
typedef __attribute__((ext_vector_type(4)))  float  v4f;
typedef __attribute__((ext_vector_type(4)))  unsigned short v4u;

#define CIN   768
#define COUT  2304
#define RNK   16

union Frag {
  v16bf v;
  v8bf  h[2];
  unsigned short s[16];
};

__device__ __forceinline__ unsigned short f2bf(float f) {
  unsigned int u = __float_as_uint(f);
  u += 0x7FFFu + ((u >> 16) & 1u);          // round-to-nearest-even
  return (unsigned short)(u >> 16);
}

// Async DMA: global (saddr base + 32-bit byte offset) -> LDS (byte offset).
// Tracked by ASYNCcnt; lower 32 bits of a generic LDS pointer are the LDS
// byte offset (flat shared aperture = {shared_base, offset}, ISA 10.2).
__device__ __forceinline__ void async_b128(unsigned lds_off, unsigned gbyte,
                                           const unsigned short* base) {
  asm volatile("global_load_async_to_lds_b128 %0, %1, %2"
               :
               : "v"(lds_off), "v"(gbyte), "s"(base)
               : "memory");
}

__device__ __forceinline__ void wait_asynccnt0() {
  asm volatile("s_wait_asynccnt 0x0" ::: "memory");
}

__device__ __forceinline__ void wait_dscnt0() {
  asm volatile("s_wait_dscnt 0x0" ::: "memory");
}

// ---------------------------------------------------------------- conversions
__global__ __launch_bounds__(256) void k_f2bf4(const float* __restrict__ in,
                                               unsigned short* __restrict__ outp,
                                               int n4) {
  int i = blockIdx.x * 256 + threadIdx.x;
  if (i >= n4) return;
  v4f v = ((const v4f*)in)[i];
  v4u r;
  r.x = f2bf(v.x); r.y = f2bf(v.y); r.z = f2bf(v.z); r.w = f2bf(v.w);
  ((v4u*)outp)[i] = r;
}

__global__ __launch_bounds__(256) void k_pad_lorab(const float* __restrict__ lb,
                                                   unsigned short* __restrict__ outp) {
  int i = blockIdx.x * 256 + threadIdx.x;      // over 768*32
  if (i >= 768 * 32) return;
  int o = i >> 5, r = i & 31;
  outp[i] = (r < RNK) ? f2bf(lb[o * RNK + r]) : (unsigned short)0;
}

__global__ __launch_bounds__(256) void k_zero(float* __restrict__ p, int n) {
  int i = blockIdx.x * 256 + threadIdx.x;
  if (i < n) p[i] = 0.0f;
}

// ------------------------------------------ x->bf16 + LoRA down-proj (WMMA)
__global__ __launch_bounds__(256) void k_xconv_lora(
    const float* __restrict__ x,
    const unsigned short* __restrict__ aq,   // bf16 [16,768]
    const unsigned short* __restrict__ av,   // bf16 [16,768]
    unsigned short* __restrict__ xb,         // bf16 [65536,768]
    float* __restrict__ dq, float* __restrict__ dv,   // f32 [65536,16]
    float* __restrict__ gq, float* __restrict__ gv)   // f32 [16,16] gate sums
{
  const int tid  = threadIdx.x;
  const int wv   = tid >> 5;
  const int lane = tid & 31;
  const int l    = lane & 15;
  const int hi   = lane >> 4;
  const long m0  = (long)blockIdx.x * 128 + (long)wv * 16;
  const long row = m0 + l;

  v8f accq = {0,0,0,0,0,0,0,0};
  v8f accv = {0,0,0,0,0,0,0,0};
  const float* xr = x + row * CIN;

  for (int k0 = 0; k0 < CIN; k0 += 32) {
    const int c0 = k0 + hi * 8;        // A-frag chunk 0 (VGPR0-3)
    const int c1 = k0 + 16 + hi * 8;   // A-frag chunk 1 (VGPR4-7)
    v4f a0 = *(const v4f*)(xr + c0);
    v4f a1 = *(const v4f*)(xr + c0 + 4);
    v4f b0 = *(const v4f*)(xr + c1);
    v4f b1 = *(const v4f*)(xr + c1 + 4);
    Frag A;
#pragma unroll
    for (int i = 0; i < 4; ++i) {
      A.s[i]      = f2bf(a0[i]);
      A.s[4 + i]  = f2bf(a1[i]);
      A.s[8 + i]  = f2bf(b0[i]);
      A.s[12 + i] = f2bf(b1[i]);
    }
    *(v8bf*)(xb + row * CIN + c0) = A.h[0];
    *(v8bf*)(xb + row * CIN + c1) = A.h[1];

    Frag Bq, Bv;
    const unsigned short* aqp = aq + (long)l * CIN + k0 + hi * 8;
    const unsigned short* avp = av + (long)l * CIN + k0 + hi * 8;
    Bq.h[0] = *(const v8bf*)(aqp);  Bq.h[1] = *(const v8bf*)(aqp + 16);
    Bv.h[0] = *(const v8bf*)(avp);  Bv.h[1] = *(const v8bf*)(avp + 16);

    accq = __builtin_amdgcn_wmma_f32_16x16x32_bf16(
        false, A.v, false, Bq.v, (short)0, accq, false, false);
    accv = __builtin_amdgcn_wmma_f32_16x16x32_bf16(
        false, A.v, false, Bv.v, (short)0, accv, false, false);
  }

  float sq = 0.0f, sv = 0.0f;
#pragma unroll
  for (int j = 0; j < 8; ++j) {            // D-layout: lane holds col l, rows hi*8+j
    const long r = m0 + hi * 8 + j;
    dq[r * RNK + l] = accq[j];
    dv[r * RNK + l] = accv[j];
    sq += accq[j];
    sv += accv[j];
  }
  const int b = (int)(m0 >> 12);           // 4096 pixels per batch
  atomicAdd(gq + b * RNK + l, sq);
  atomicAdd(gv + b * RNK + l, sv);
}

// ------------------------------------------------------------------- gating
__global__ void k_gate(const float* __restrict__ gacc,
                       const float* __restrict__ gw,
                       const float* __restrict__ gb,
                       float* __restrict__ probs) {
  const int b = threadIdx.x;
  if (b >= 16) return;
  float g[16];
#pragma unroll
  for (int r = 0; r < 16; ++r) g[r] = gacc[b * 16 + r] * (1.0f / 4096.0f);
  float lg[3];
  float mx = -1e30f;
#pragma unroll
  for (int e = 0; e < 3; ++e) {
    float s = gb[e];
#pragma unroll
    for (int r = 0; r < 16; ++r) s += g[r] * gw[e * 16 + r];
    lg[e] = s;
    mx = fmaxf(mx, s);
  }
  float den = 0.0f, ex[3];
#pragma unroll
  for (int e = 0; e < 3; ++e) { ex[e] = __expf(lg[e] - mx); den += ex[e]; }
#pragma unroll
  for (int e = 0; e < 3; ++e) probs[b * 3 + e] = ex[e] / den;
}

// -------------------------------------------- experts (dw + pw, prob-mixed)
#define TS   16
#define HALO 3
#define TD   22   // TS + 2*HALO

__global__ __launch_bounds__(256) void k_experts(
    const float* __restrict__ dq, const float* __restrict__ dv,
    const float* __restrict__ probsq, const float* __restrict__ probsv,
    const float* __restrict__ dwq0, const float* __restrict__ dwq1, const float* __restrict__ dwq2,
    const float* __restrict__ pwq0, const float* __restrict__ pwq1, const float* __restrict__ pwq2,
    const float* __restrict__ dwv0, const float* __restrict__ dwv1, const float* __restrict__ dwv2,
    const float* __restrict__ pwv0, const float* __restrict__ pwv1, const float* __restrict__ pwv2,
    unsigned short* __restrict__ mq, unsigned short* __restrict__ mv)   // bf16 [65536,32]
{
  __shared__ float sq[TD * TD * 16];
  __shared__ float sv[TD * TD * 16];
  const int tid = threadIdx.x;
  const int x0 = blockIdx.x * TS, y0 = blockIdx.y * TS, b = blockIdx.z;
  const v4f z = {0, 0, 0, 0};

  for (int idx = tid; idx < TD * TD; idx += 256) {
    int py = idx / TD, px = idx % TD;
    int gy = y0 + py - HALO, gx = x0 + px - HALO;
    bool ok = (gy >= 0) & (gy < 64) & (gx >= 0) & (gx < 64);
    long base = (((long)b * 64 + gy) * 64 + gx) * 16;
#pragma unroll
    for (int r = 0; r < 16; r += 4) {
      v4f vq = z, vv = z;
      if (ok) {
        vq = *(const v4f*)(dq + base + r);
        vv = *(const v4f*)(dv + base + r);
      }
      *(v4f*)(sq + idx * 16 + r) = vq;
      *(v4f*)(sv + idx * 16 + r) = vv;
    }
  }
  __syncthreads();

  const int ty = tid >> 4, tx = tid & 15;
  const float* dwq[3] = {dwq0, dwq1, dwq2};
  const float* pwq[3] = {pwq0, pwq1, pwq2};
  const float* dwv[3] = {dwv0, dwv1, dwv2};
  const float* pwv[3] = {pwv0, pwv1, pwv2};

  float accq[16], accv[16];
#pragma unroll
  for (int r = 0; r < 16; ++r) { accq[r] = 0.0f; accv[r] = 0.0f; }

#pragma unroll
  for (int e = 0; e < 3; ++e) {
    const int ksz = 3 + 2 * e;
    const int off = HALO - (ksz >> 1);
    float yq[16], yv[16];
#pragma unroll
    for (int r = 0; r < 16; ++r) { yq[r] = 0.0f; yv[r] = 0.0f; }
    for (int i = 0; i < ksz; ++i) {
      for (int j = 0; j < ksz; ++j) {
        const float* cq = sq + ((ty + off + i) * TD + (tx + off + j)) * 16;
        const float* cv = sv + ((ty + off + i) * TD + (tx + off + j)) * 16;
        const float* wq = dwq[e] + (i * ksz + j) * 16;
        const float* wvp = dwv[e] + (i * ksz + j) * 16;
#pragma unroll
        for (int r = 0; r < 16; ++r) {
          yq[r] += cq[r] * wq[r];
          yv[r] += cv[r] * wvp[r];
        }
      }
    }
    const float pq = probsq[b * 3 + e];
    const float pv = probsv[b * 3 + e];
#pragma unroll
    for (int o = 0; o < 16; ++o) {
      float zq = 0.0f, zv = 0.0f;
#pragma unroll
      for (int r = 0; r < 16; ++r) {
        zq += yq[r] * pwq[e][r * 16 + o];   // pw: [1,1,r_in,r_out]
        zv += yv[r] * pwv[e][r * 16 + o];
      }
      accq[o] += pq * zq;
      accv[o] += pv * zv;
    }
  }

  const long m = ((long)b * 64 + (y0 + ty)) * 64 + (x0 + tx);
#pragma unroll
  for (int r = 0; r < 16; ++r) {
    mq[m * 32 + r] = f2bf(accq[r]);
    mv[m * 32 + r] = f2bf(accv[r]);
  }
#pragma unroll
  for (int r = 16; r < 32; ++r) { mq[m * 32 + r] = 0; mv[m * 32 + r] = 0; }
}

// ---------------------------------------------- main fused QKV GEMM (WMMA)
// Block = 256 threads (8 waves). Block tile: 128 rows x 128 cols.
// W k-slices (128x32 bf16 = 8 KB) staged into double-buffered LDS with
// GLOBAL_LOAD_ASYNC_TO_LDS_B128 (ASYNCcnt), copy of tile k+1 overlapped
// with WMMAs on tile k; one barrier per k-iteration.
__global__ __launch_bounds__(256) void k_qkv_gemm(
    const unsigned short* __restrict__ xb,   // bf16 [65536,768]
    const unsigned short* __restrict__ Wb,   // bf16 [2304,768]
    const float* __restrict__ bias,          // f32 [2304]
    const unsigned short* __restrict__ mq,   // bf16 [65536,32] (K padded)
    const unsigned short* __restrict__ mv,
    const unsigned short* __restrict__ lbq,  // bf16 [768,32]  (K padded)
    const unsigned short* __restrict__ lbv,
    float* __restrict__ out)                 // f32 [65536,2304]
{
  __shared__ unsigned short lB[2][128 * 40]; // 2 x (128 rows x 32 k, pad->40)
  __shared__ float sOut[8][16 * 132];        // per-wave store-transpose tile
  const int tid  = threadIdx.x;
  const int wv   = tid >> 5;
  const int lane = tid & 31;
  const int l    = lane & 15;
  const int hi   = lane >> 4;
  const long m0  = (long)blockIdx.x * 128 + (long)wv * 16;
  const int  n0  = blockIdx.y * 128;
  const long rowA = m0 + l;

  v8f acc[8];
#pragma unroll
  for (int t = 0; t < 8; ++t) {
    const float bvv = bias[n0 + t * 16 + l]; // C-layout: lane holds column l
    v8f tmp = {bvv, bvv, bvv, bvv, bvv, bvv, bvv, bvv};
    acc[t] = tmp;
  }

  // async staging setup: thread t covers W row n0+sn, 32-byte half skb.
  const unsigned ldsW = (unsigned)(unsigned long long)(void*)&lB[0][0];
  const int sn  = tid >> 1;
  const int skb = (tid & 1) * 32;                               // bytes
  const unsigned grow = (unsigned)((n0 + sn) * CIN) * 2u + (unsigned)skb;
  const unsigned ldst = ldsW + (unsigned)(sn * 80 + skb);

  // prologue: stage k0 = 0 into buffer 0
  async_b128(ldst,       grow,       Wb);
  async_b128(ldst + 16u, grow + 16u, Wb);
  wait_asynccnt0();
  __syncthreads();

  for (int k0 = 0; k0 < CIN; k0 += 32) {
    const int cur = (k0 >> 5) & 1;
    if (k0 + 32 < CIN) {                     // stage next tile (overlapped)
      const unsigned dst = ldst + (unsigned)((cur ^ 1) * (128 * 40 * 2));
      const unsigned src = grow + (unsigned)((k0 + 32) * 2);
      async_b128(dst,       src,       Wb);
      async_b128(dst + 16u, src + 16u, Wb);
    }
    __builtin_prefetch(xb + rowA * CIN + k0 + 64, 0, 1);  // global_prefetch_b8

    Frag A;
    A.h[0] = *(const v8bf*)(xb + rowA * CIN + k0 + hi * 8);
    A.h[1] = *(const v8bf*)(xb + rowA * CIN + k0 + 16 + hi * 8);
#pragma unroll
    for (int t = 0; t < 8; ++t) {
      Frag Bf;
      const unsigned short* bp = &lB[cur][(t * 16 + l) * 40 + hi * 8];
      Bf.h[0] = *(const v8bf*)(bp);
      Bf.h[1] = *(const v8bf*)(bp + 16);
      acc[t] = __builtin_amdgcn_wmma_f32_16x16x32_bf16(
          false, A.v, false, Bf.v, (short)0, acc[t], false, false);
    }
    wait_asynccnt0();                        // next tile landed in LDS
    __syncthreads();
  }

  // Fused LoRA up-projection epilogue (q third: y<6, v third: y>=12).
  if (blockIdx.y < 6 || blockIdx.y >= 12) {
    const bool isq = (blockIdx.y < 6);
    const unsigned short* mix = isq ? mq : mv;
    const unsigned short* lb  = isq ? lbq : lbv;
    const int o0 = n0 - (isq ? 0 : 1536);
    Frag Am;
    Am.h[0] = *(const v8bf*)(mix + rowA * 32 + hi * 8);
    Am.h[1] = *(const v8bf*)(mix + rowA * 32 + 16 + hi * 8);
#pragma unroll
    for (int t = 0; t < 8; ++t) {
      Frag Bm;
      const unsigned short* bp = lb + (long)(o0 + t * 16 + l) * 32 + hi * 8;
      Bm.h[0] = *(const v8bf*)(bp);
      Bm.h[1] = *(const v8bf*)(bp + 16);
      acc[t] = __builtin_amdgcn_wmma_f32_16x16x32_bf16(
          false, Am.v, false, Bm.v, (short)0, acc[t], false, false);
    }
  }

  // Transpose through per-wave LDS tile so stores are contiguous b128s.
  float* so = &sOut[wv][0];
#pragma unroll
  for (int t = 0; t < 8; ++t)
#pragma unroll
    for (int j = 0; j < 8; ++j)
      so[(hi * 8 + j) * 132 + t * 16 + l] = acc[t][j];
  wait_dscnt0();                             // intra-wave LDS visibility

  const int rr   = lane & 15;                // row within wave tile
  const int half = lane >> 4;                // column half (0-63 / 64-127)
  const float* srow = so + rr * 132 + half * 64;
  float* orow = out + (m0 + rr) * COUT + n0 + half * 64;
#pragma unroll
  for (int i = 0; i < 16; ++i) {
    v4f vv4 = *(const v4f*)(srow + i * 4);
    *(v4f*)(orow + i * 4) = vv4;
  }
}

// --------------------------------------------------------------------------
extern "C" void kernel_launch(void* const* d_in, const int* in_sizes, int n_in,
                              void* d_out, int out_size, void* d_ws, size_t ws_size,
                              hipStream_t stream) {
  (void)in_sizes; (void)n_in; (void)out_size; (void)ws_size;

  const float* x    = (const float*)d_in[0];
  const float* W    = (const float*)d_in[1];
  const float* bias = (const float*)d_in[2];
  const float* laq  = (const float*)d_in[3];
  const float* lbq  = (const float*)d_in[4];
  const float* lav  = (const float*)d_in[5];
  const float* lbv  = (const float*)d_in[6];
  const float* gqw  = (const float*)d_in[7];
  const float* gqb  = (const float*)d_in[8];
  const float* gvw  = (const float*)d_in[9];
  const float* gvb  = (const float*)d_in[10];
  const float* dwq0 = (const float*)d_in[11];
  const float* pwq0 = (const float*)d_in[12];
  const float* dwv0 = (const float*)d_in[13];
  const float* pwv0 = (const float*)d_in[14];
  const float* dwq1 = (const float*)d_in[15];
  const float* pwq1 = (const float*)d_in[16];
  const float* dwv1 = (const float*)d_in[17];
  const float* pwv1 = (const float*)d_in[18];
  const float* dwq2 = (const float*)d_in[19];
  const float* pwq2 = (const float*)d_in[20];
  const float* dwv2 = (const float*)d_in[21];
  const float* pwv2 = (const float*)d_in[22];
  float* out = (float*)d_out;

  char* ws = (char*)d_ws;
  // workspace layout (bytes)
  constexpr size_t OFF_XB = 0;            // bf16 x          100,663,296
  constexpr size_t OFF_WB = 100663296;    // bf16 W            3,538,944
  constexpr size_t OFF_AQ = 104202240;    // bf16 lora_a_q        24,576
  constexpr size_t OFF_AV = 104226816;    // bf16 lora_a_v        24,576
  constexpr size_t OFF_BQ = 104251392;    // bf16 lora_b_q pad    49,152
  constexpr size_t OFF_BV = 104300544;    // bf16 lora_b_v pad    49,152
  constexpr size_t OFF_DQ = 104349696;    // f32 d_q           4,194,304
  constexpr size_t OFF_DV = 108544000;    // f32 d_v           4,194,304
  constexpr size_t OFF_MQ = 112738304;    // bf16 mixed_q pad  4,194,304
  constexpr size_t OFF_MV = 116932608;    // bf16 mixed_v pad  4,194,304
  constexpr size_t OFF_GQ = 121126912;    // f32 gate acc q        1,024
  constexpr size_t OFF_GV = 121127936;    // f32 gate acc v        1,024
  constexpr size_t OFF_PQ = 121128960;    // f32 probs q             256
  constexpr size_t OFF_PV = 121129216;    // f32 probs v             256

  unsigned short* xb  = (unsigned short*)(ws + OFF_XB);
  unsigned short* Wb  = (unsigned short*)(ws + OFF_WB);
  unsigned short* aqb = (unsigned short*)(ws + OFF_AQ);
  unsigned short* avb = (unsigned short*)(ws + OFF_AV);
  unsigned short* bqp = (unsigned short*)(ws + OFF_BQ);
  unsigned short* bvp = (unsigned short*)(ws + OFF_BV);
  float* dq = (float*)(ws + OFF_DQ);
  float* dv = (float*)(ws + OFF_DV);
  unsigned short* mqb = (unsigned short*)(ws + OFF_MQ);
  unsigned short* mvb = (unsigned short*)(ws + OFF_MV);
  float* gq = (float*)(ws + OFF_GQ);
  float* gv = (float*)(ws + OFF_GV);
  float* pq = (float*)(ws + OFF_PQ);
  float* pv = (float*)(ws + OFF_PV);

  // 1) operand conversions
  k_f2bf4<<<dim3(1728), 256, 0, stream>>>(W, Wb, 2304 * 768 / 4);
  k_f2bf4<<<dim3(12),   256, 0, stream>>>(laq, aqb, 16 * 768 / 4);
  k_f2bf4<<<dim3(12),   256, 0, stream>>>(lav, avb, 16 * 768 / 4);
  k_pad_lorab<<<dim3(96), 256, 0, stream>>>(lbq, bqp);
  k_pad_lorab<<<dim3(96), 256, 0, stream>>>(lbv, bvp);
  k_zero<<<dim3(2), 256, 0, stream>>>(gq, 512);   // gq + gv contiguous

  // 2) x -> bf16 + LoRA down-projection + gate partials (WMMA)
  k_xconv_lora<<<dim3(512), 256, 0, stream>>>(x, aqb, avb, xb, dq, dv, gq, gv);

  // 3) softmax gates
  k_gate<<<dim3(1), 32, 0, stream>>>(gq, gqw, gqb, pq);
  k_gate<<<dim3(1), 32, 0, stream>>>(gv, gvw, gvb, pv);

  // 4) experts + mixing -> padded bf16
  k_experts<<<dim3(4, 4, 16), 256, 0, stream>>>(
      dq, dv, pq, pv,
      dwq0, dwq1, dwq2, pwq0, pwq1, pwq2,
      dwv0, dwv1, dwv2, pwv0, pwv1, pwv2,
      mqb, mvb);

  // 5) main fused GEMM (M=65536, N=2304, K=768) + LoRA epilogue
  k_qkv_gemm<<<dim3(512, 18), 256, 0, stream>>>(
      xb, Wb, bias, mqb, mvb, bqp, bvp, out);
}
